// MultiHeadAttention_48017734369681
// MI455X (gfx1250) — matrix operand order repaired and checked by reading, not compiled, not run on hardware
//
#include <hip/hip_runtime.h>
#include <math.h>

// ---------------------------------------------------------------------------
// MHA for MI455X (gfx1250, wave32, WMMA).  All heavy math runs through
// v_wmma_f32_16x16x32_bf16 (fp32 inputs converted to bf16, f32 accumulate).
// Pipeline: cvt -> GEMM(Q) -> GEMM(K) -> GEMM(V, stored transposed) ->
//           flash-attention -> GEMM(out, fp32).
// GEMM uses a register-staged double-buffered LDS pipeline (1 barrier/iter);
// attention prefetches the next K/V block and batches its cross-lane
// softmax reductions step-major so 8 ds_bpermutes fly per wait.
// ---------------------------------------------------------------------------

#define EMBD   1024
#define HEADS  16
#define DHEAD  64
#define SEQ    2048
#define BATCH  4
#define TOKENS (BATCH * SEQ)   // 8192

typedef __bf16 bf16_t;
typedef __attribute__((ext_vector_type(16))) __bf16 v16bf;
typedef __attribute__((ext_vector_type(8)))  __bf16 v8bf;
typedef __attribute__((ext_vector_type(4)))  __bf16 v4bf;
typedef __attribute__((ext_vector_type(8)))  float  v8f;
typedef __attribute__((ext_vector_type(4)))  float  v4f;

static __device__ __forceinline__ v16bf cat8(v8bf lo, v8bf hi) {
  return __builtin_shufflevector(lo, hi, 0,1,2,3,4,5,6,7,8,9,10,11,12,13,14,15);
}
// A-operand (16x32 bf16): lane half kgrp -> K chunks at {kgrp*8, kgrp*8+16}.
// Caller passes p already offset by kgrp*8; we fetch +0 and +16 elements.
static __device__ __forceinline__ v16bf frag_a(const bf16_t* p) {
  return cat8(*(const v8bf*)p, *(const v8bf*)(p + 16));
}
// B-operand (32x16 bf16): lane half kgrp -> K chunks at {kgrp*8, kgrp*8+16}.
// Caller passes p already offset by hl*8; we fetch +0 and +16 elements.
static __device__ __forceinline__ v16bf frag_b(const bf16_t* p) {
  return cat8(*(const v8bf*)p, *(const v8bf*)(p + 16));
}
static __device__ __forceinline__ v8f wmma_bf16(v16bf a, v16bf b, v8f c) {
  return __builtin_amdgcn_wmma_f32_16x16x32_bf16(false, a, false, b,
                                                 (short)0, c, false, false);
}

// ---------------------------------------------------------------------------
// fp32 -> bf16 conversion (vectorized x4)
// ---------------------------------------------------------------------------
__global__ void cvt_f32_bf16(const float* __restrict__ src,
                             bf16_t* __restrict__ dst, int n4) {
  int i = blockIdx.x * blockDim.x + threadIdx.x;
  if (i < n4) {
    v4f x = ((const v4f*)src)[i];
    ((v4bf*)dst)[i] = __builtin_convertvector(x, v4bf);
  }
}

// ---------------------------------------------------------------------------
// GEMM: Y[M=8192, N=1024] = A[8192,1024](bf16) * W[1024,1024]^T(bf16) + bias
// Block tile 128x128, 8 waves (2x4), wave tile 64x32, K staged 32 at a time.
// Register-staged double-buffered LDS (padded stride 40 -> conflict-free):
//   iter k: store regs->LDS[k&1]; issue global loads for k+1; barrier;
//           WMMA from LDS[k&1].
// MODE 0: store bf16 split-heads [b,h,n,d]      (Q, K)
// MODE 1: store bf16 transposed  [b,h,d,n]      (V)
// MODE 2: store fp32 plain       [token, e]     (final output)
// ---------------------------------------------------------------------------
#define LDT 40
#define NKS (EMBD / 32)   // 32 K-steps

template <int MODE>
__global__ __launch_bounds__(256, 1)
void gemm_xwT(const bf16_t* __restrict__ A, const bf16_t* __restrict__ W,
              const float* __restrict__ bias, void* __restrict__ out) {
  __shared__ __align__(32) bf16_t sA[2][128 * LDT];
  __shared__ __align__(32) bf16_t sB[2][128 * LDT];

  const int tid  = threadIdx.x;
  const int lane = tid & 31;
  const int wave = tid >> 5;
  const int wm   = wave >> 2;   // 0..1
  const int wn   = wave & 3;    // 0..3
  const int lr   = lane & 15;
  const int hl   = lane >> 4;
  const int m0   = blockIdx.y * 128;
  const int n0   = blockIdx.x * 128;

  v8f acc[4][2];
#pragma unroll
  for (int i = 0; i < 4; i++)
#pragma unroll
    for (int j = 0; j < 2; j++) acc[i][j] = (v8f){0.f,0.f,0.f,0.f,0.f,0.f,0.f,0.f};

  // loader: each thread owns 2x16B chunks of the A tile and of the B tile
  const int c0   = tid * 2;
  const int row0 = (c0 + 0) >> 2, seg0 = ((c0 + 0) & 3) * 8;
  const int row1 = (c0 + 1) >> 2, seg1 = ((c0 + 1) & 3) * 8;
  const bf16_t* gA0 = &A[(size_t)(m0 + row0) * EMBD + seg0];
  const bf16_t* gA1 = &A[(size_t)(m0 + row1) * EMBD + seg1];
  const bf16_t* gB0 = &W[(size_t)(n0 + row0) * EMBD + seg0];
  const bf16_t* gB1 = &W[(size_t)(n0 + row1) * EMBD + seg1];

  v8bf ra0, ra1, rb0, rb1;
  ra0 = *(const v8bf*)(gA0);  ra1 = *(const v8bf*)(gA1);
  rb0 = *(const v8bf*)(gB0);  rb1 = *(const v8bf*)(gB1);

  for (int kb = 0; kb < NKS; ++kb) {
    const int buf = kb & 1;
    // stage current slice into LDS
    *(v8bf*)(&sA[buf][row0 * LDT + seg0]) = ra0;
    *(v8bf*)(&sA[buf][row1 * LDT + seg1]) = ra1;
    *(v8bf*)(&sB[buf][row0 * LDT + seg0]) = rb0;
    *(v8bf*)(&sB[buf][row1 * LDT + seg1]) = rb1;
    // issue global loads for the next slice before the barrier
    if (kb + 1 < NKS) {
      const int k1 = (kb + 1) * 32;
      ra0 = *(const v8bf*)(gA0 + k1);  ra1 = *(const v8bf*)(gA1 + k1);
      rb0 = *(const v8bf*)(gB0 + k1);  rb1 = *(const v8bf*)(gB1 + k1);
    }
    __syncthreads();

    v16bf af[4], bfr[2];
#pragma unroll
    for (int i = 0; i < 4; i++)
      af[i] = frag_a(&sA[buf][(wm * 64 + i * 16 + lr) * LDT + hl * 8]);
#pragma unroll
    for (int j = 0; j < 2; j++)
      bfr[j] = frag_b(&sB[buf][(wn * 32 + j * 16 + lr) * LDT + hl * 8]);
#pragma unroll
    for (int i = 0; i < 4; i++)
#pragma unroll
      for (int j = 0; j < 2; j++)
        acc[i][j] = wmma_bf16(af[i], bfr[j], acc[i][j]);
  }

  // epilogue: bias + layout-specific store
#pragma unroll
  for (int i = 0; i < 4; i++) {
#pragma unroll
    for (int j = 0; j < 2; j++) {
      const int ng = n0 + wn * 32 + j * 16 + lr;
      const float bv = bias[ng];
#pragma unroll
      for (int r = 0; r < 8; r++) {
        const int   mg  = m0 + wm * 64 + i * 16 + r + hl * 8;
        const float val = acc[i][j][r] + bv;
        if (MODE == 2) {
          ((float*)out)[(size_t)mg * EMBD + ng] = val;
        } else {
          const int b = mg >> 11, nt = mg & (SEQ - 1);
          const int h = ng >> 6,  d  = ng & (DHEAD - 1);
          const size_t dst =
              (MODE == 0)
                  ? ((size_t)((b * HEADS + h) * SEQ + nt)) * DHEAD + d
                  : ((size_t)((b * HEADS + h) * DHEAD + d)) * SEQ + nt;
          ((bf16_t*)out)[dst] = (bf16_t)val;
        }
      }
    }
  }
}

// ---------------------------------------------------------------------------
// Flash attention: 1 block = 64 query rows of one (b,h); 4 waves x 16 rows.
// S = Q*K^T (B operand = K rows, contiguous), online softmax in C layout,
// P re-shaped via per-wave padded LDS strip into A layout, O += P*V with V
// pre-transposed so B operand reads are contiguous.  scale = 1/sqrt(EMBD).
// Cross-lane reductions are batched step-major across the 8 rows.
// ---------------------------------------------------------------------------
#define PLD 72  // padded P strip stride (halfwords) -> conflict-free reads

__global__ __launch_bounds__(128, 1)
void attention_kernel(const bf16_t* __restrict__ Q,   // [B,H,SEQ,DHEAD]
                      const bf16_t* __restrict__ K,   // [B,H,SEQ,DHEAD]
                      const bf16_t* __restrict__ Vt,  // [B,H,DHEAD,SEQ]
                      bf16_t* __restrict__ O) {       // [TOKENS,EMBD]
  const int lane = threadIdx.x & 31;
  const int wave = threadIdx.x >> 5;  // 0..3
  const int lr   = lane & 15;
  const int hl   = lane >> 4;
  const int qt   = blockIdx.x & 31;
  const int bh   = blockIdx.x >> 5;   // 0..63
  const int b    = bh >> 4;
  const int h    = bh & 15;
  const int q0   = qt * 64 + wave * 16;

  __shared__ __align__(32) bf16_t sP[4][16 * PLD];
  bf16_t* Pw = &sP[wave][0];

  const bf16_t* Qb = Q  + (size_t)bh * SEQ * DHEAD;
  const bf16_t* Kb = K  + (size_t)bh * SEQ * DHEAD;
  const bf16_t* Vb = Vt + (size_t)bh * DHEAD * SEQ;

  // Q fragments (A layout), loaded once: rows q0+lr, d in [t*32, t*32+32)
  v16bf aq[2];
#pragma unroll
  for (int t = 0; t < 2; t++)
    aq[t] = frag_a(&Qb[(size_t)(q0 + lr) * DHEAD + t * 32 + hl * 8]);

  v8f o[4];
#pragma unroll
  for (int j = 0; j < 4; j++) o[j] = (v8f){0.f,0.f,0.f,0.f,0.f,0.f,0.f,0.f};
  float mrun[8], lsum[8];
#pragma unroll
  for (int r = 0; r < 8; r++) { mrun[r] = -__builtin_inff(); lsum[r] = 0.f; }

  const float CS = 1.4426950408889634f / 32.0f;  // log2(e)/sqrt(EMBD)

  for (int kb = 0; kb < SEQ; kb += 64) {
    // prefetch next key block (K rows) and next V^T column segment
    if (kb + 64 < SEQ) {
      __builtin_prefetch(&Kb[(size_t)(kb + 64 + lane) * DHEAD], 0, 3);
      __builtin_prefetch(&Kb[(size_t)(kb + 96 + lane) * DHEAD], 0, 3);
      __builtin_prefetch(&Vb[(size_t)lane * SEQ + kb + 64], 0, 3);
      __builtin_prefetch(&Vb[(size_t)(lane + 32) * SEQ + kb + 64], 0, 3);
    }
    // ---- S = Q * K^T (16 x 64 per wave) ----
    v8f s[4];
#pragma unroll
    for (int j = 0; j < 4; j++) {
      v8f sv = (v8f){0.f,0.f,0.f,0.f,0.f,0.f,0.f,0.f};
#pragma unroll
      for (int t = 0; t < 2; t++) {
        v16bf bk = frag_b(&Kb[(size_t)(kb + j * 16 + lr) * DHEAD +
                              t * 32 + hl * 8]);
        sv = wmma_bf16(aq[t], bk, sv);
      }
      s[j] = sv;
    }
    // ---- online softmax: batched step-major cross-lane reductions ----
    float mx[8], ps[8], al[8];
#pragma unroll
    for (int r = 0; r < 8; r++)
      mx[r] = fmaxf(fmaxf(s[0][r], s[1][r]), fmaxf(s[2][r], s[3][r]));
#pragma unroll
    for (int m = 1; m < 16; m <<= 1)
#pragma unroll
      for (int r = 0; r < 8; r++)
        mx[r] = fmaxf(mx[r], __shfl_xor(mx[r], m, 32));
#pragma unroll
    for (int r = 0; r < 8; r++) {
      const float mnew = fmaxf(mrun[r], mx[r]);
      al[r] = exp2f((mrun[r] - mnew) * CS);
      mrun[r] = mnew;
      float p0 = exp2f((s[0][r] - mnew) * CS);
      float p1 = exp2f((s[1][r] - mnew) * CS);
      float p2 = exp2f((s[2][r] - mnew) * CS);
      float p3 = exp2f((s[3][r] - mnew) * CS);
      s[0][r] = p0; s[1][r] = p1; s[2][r] = p2; s[3][r] = p3;
      ps[r] = (p0 + p1) + (p2 + p3);
    }
#pragma unroll
    for (int m = 1; m < 16; m <<= 1)
#pragma unroll
      for (int r = 0; r < 8; r++)
        ps[r] += __shfl_xor(ps[r], m, 32);
#pragma unroll
    for (int r = 0; r < 8; r++) {
      lsum[r] = lsum[r] * al[r] + ps[r];
#pragma unroll
      for (int j = 0; j < 4; j++) o[j][r] *= al[r];
    }
    // ---- P (C layout) -> LDS strip (row-major 16x64, stride PLD) ----
    __syncthreads();
#pragma unroll
    for (int j = 0; j < 4; j++)
#pragma unroll
      for (int r = 0; r < 8; r++)
        Pw[(r + hl * 8) * PLD + j * 16 + lr] = (bf16_t)s[j][r];
    __syncthreads();
    // ---- O += P * V ----
#pragma unroll
    for (int t = 0; t < 2; t++) {
      v16bf pa = frag_a(&Pw[lr * PLD + t * 32 + hl * 8]);
#pragma unroll
      for (int j = 0; j < 4; j++) {
        v16bf bv = frag_b(&Vb[(size_t)(j * 16 + lr) * SEQ + kb +
                              t * 32 + hl * 8]);
        o[j] = wmma_bf16(pa, bv, o[j]);
      }
    }
  }

  // ---- normalize & store merged-head output [token, h*64+d] (bf16) ----
#pragma unroll
  for (int r = 0; r < 8; r++) {
    const float inv = 1.0f / lsum[r];
    const int tok = b * SEQ + q0 + r + hl * 8;
#pragma unroll
    for (int j = 0; j < 4; j++) {
      O[(size_t)tok * EMBD + h * DHEAD + j * 16 + lr] =
          (bf16_t)(o[j][r] * inv);
    }
  }
}

// ---------------------------------------------------------------------------
// Launcher
// ---------------------------------------------------------------------------
extern "C" void kernel_launch(void* const* d_in, const int* in_sizes, int n_in,
                              void* d_out, int out_size, void* d_ws,
                              size_t ws_size, hipStream_t stream) {
  (void)in_sizes; (void)n_in; (void)out_size; (void)ws_size;
  const float* X  = (const float*)d_in[0];
  const float* Wq = (const float*)d_in[1];
  const float* bq = (const float*)d_in[2];
  const float* Wk = (const float*)d_in[3];
  const float* bk = (const float*)d_in[4];
  const float* Wv = (const float*)d_in[5];
  const float* bv = (const float*)d_in[6];
  const float* Wo = (const float*)d_in[7];
  const float* bo = (const float*)d_in[8];
  float* out = (float*)d_out;

  char* ws = (char*)d_ws;
  const size_t SZ_X = (size_t)TOKENS * EMBD * sizeof(bf16_t);  // 16 MB
  const size_t SZ_W = (size_t)EMBD * EMBD * sizeof(bf16_t);    //  2 MB
  bf16_t* Xb  = (bf16_t*)(ws);
  bf16_t* Wqb = (bf16_t*)(ws + SZ_X);
  bf16_t* Wkb = (bf16_t*)(ws + SZ_X + SZ_W);
  bf16_t* Wvb = (bf16_t*)(ws + SZ_X + 2 * SZ_W);
  bf16_t* Wob = (bf16_t*)(ws + SZ_X + 3 * SZ_W);
  bf16_t* Qb  = (bf16_t*)(ws + SZ_X + 4 * SZ_W);
  bf16_t* Kb  = (bf16_t*)(ws + 2 * SZ_X + 4 * SZ_W);
  bf16_t* Vtb = (bf16_t*)(ws + 3 * SZ_X + 4 * SZ_W);
  bf16_t* Ob  = (bf16_t*)(ws + 4 * SZ_X + 4 * SZ_W);

  // 1) fp32 -> bf16 conversions
  {
    int n4 = TOKENS * EMBD / 4;
    cvt_f32_bf16<<<(n4 + 255) / 256, 256, 0, stream>>>(X, Xb, n4);
    int w4 = EMBD * EMBD / 4;
    cvt_f32_bf16<<<(w4 + 255) / 256, 256, 0, stream>>>(Wq, Wqb, w4);
    cvt_f32_bf16<<<(w4 + 255) / 256, 256, 0, stream>>>(Wk, Wkb, w4);
    cvt_f32_bf16<<<(w4 + 255) / 256, 256, 0, stream>>>(Wv, Wvb, w4);
    cvt_f32_bf16<<<(w4 + 255) / 256, 256, 0, stream>>>(Wo, Wob, w4);
  }

  // 2) projections
  dim3 gg(EMBD / 128, TOKENS / 128);  // (8, 64)
  gemm_xwT<0><<<gg, 256, 0, stream>>>(Xb, Wqb, bq, (void*)Qb);
  gemm_xwT<0><<<gg, 256, 0, stream>>>(Xb, Wkb, bk, (void*)Kb);
  gemm_xwT<1><<<gg, 256, 0, stream>>>(Xb, Wvb, bv, (void*)Vtb);

  // 3) attention: 4*16 heads x 32 query tiles
  attention_kernel<<<BATCH * HEADS * (SEQ / 64), 128, 0, stream>>>(Qb, Kb, Vtb,
                                                                   Ob);

  // 4) output projection (fp32 result)
  gemm_xwT<2><<<gg, 256, 0, stream>>>(Ob, Wob, bo, (void*)out);
}
